// EdgeBlock_86844238725703
// MI455X (gfx1250) — matrix operand order repaired and checked by reading, not compiled, not run on hardware
//
#include <hip/hip_runtime.h>
#include <hip/hip_bf16.h>

// ---------------- problem constants ----------------
#define Bg   8
#define Nn   512
#define Eg   262144
#define NDIM 128
#define EDIM 64
#define GDIM 32
#define FIN  320
#define XD   352      // FIN + GDIM
#define H1D  256
#define OUTD 64
#define AHD  64
#define NH   4

// WMMA tiling (K=32 per v_wmma_f32_16x16x32_f16)
#define KT1 11        // XD/32
#define NT1 16        // H1D/16
#define KT2 8         // H1D/32
#define NT2 4         // OUTD/16
#define NTA 4         // AHD/16
#define KTA 2         // AHD/32

// LDS strides (f16 elements); row stride bytes must be a 16B multiple
#define XSTR 368      // 736B rows, 176+8 dw -> conflict-free-ish banking
#define HSTR 264      // 528B rows
#define ASTR 72       // 144B rows (aliased inside ht buffer)

// ---------------- workspace layout (bytes) ----------------
#define OFF_W1P 0u
#define SZ_W1P  (KT1*NT1*512u*2u)          // 180224
#define OFF_A1P (OFF_W1P + SZ_W1P)
#define SZ_A1P  (KT1*NTA*512u*2u)          // 45056
#define OFF_W2P (OFF_A1P + SZ_A1P)
#define SZ_W2P  (KT2*NT2*512u*2u)          // 32768
#define OFF_A2P (OFF_W2P + SZ_W2P)
#define SZ_A2P  (KTA*1u*512u*2u)           // 2048
#define OFF_ASC 262144u                    // f32 scores E*NH
#define SZ_ASC  ((unsigned)Eg*NH*4u)
#define OFF_SMX (OFF_ASC + SZ_ASC)
#define OFF_DEN (OFF_SMX + (unsigned)(Bg*Nn*NH)*4u)

typedef _Float16 h16;
typedef __attribute__((ext_vector_type(16))) _Float16 v16h;
typedef __attribute__((ext_vector_type(8)))  _Float16 v8h;
typedef __attribute__((ext_vector_type(8)))  float    v8f;

union U16 { v16h v; v8h h[2]; };

__device__ __forceinline__ v8f wmma16(v16h a, v16h b, v8f c) {
  return __builtin_amdgcn_wmma_f32_16x16x32_f16(
      /*neg_a=*/false, a, /*neg_b=*/false, b,
      /*c_mod=*/(short)0, c, /*reuse_a=*/false, /*reuse_b=*/false);
}

__device__ __forceinline__ float silu(float x) {
  return x / (1.f + __expf(-x));
}

// monotone float<->uint mapping for atomic max on floats
__device__ __forceinline__ unsigned fenc(float x) {
  unsigned u = __builtin_bit_cast(unsigned, x);
  return (u & 0x80000000u) ? ~u : (u | 0x80000000u);
}
__device__ __forceinline__ float fdec(unsigned e) {
  unsigned u = (e & 0x80000000u) ? (e & 0x7FFFFFFFu) : ~e;
  return __builtin_bit_cast(float, u);
}

__device__ __forceinline__ void put4(h16* d, float4 v, float& sum, float& ss) {
  sum += (v.x + v.y) + (v.z + v.w);
  ss  += (v.x * v.x + v.y * v.y) + (v.z * v.z + v.w * v.w);
  d[0] = (h16)v.x; d[1] = (h16)v.y; d[2] = (h16)v.z; d[3] = (h16)v.w;
}

// ---------------- weight packing: B-fragment layout ----------------
// fragment f = kt*NT + nt; lane holds col n = nt*16+(lane&15),
// rows k = kt*32 + (lane>>4)*8 + (j&7) + (j>>3)*16, j=0..15 (ISA 16-bit B 32x16).
__device__ __forceinline__ void pack_one(const float* M, h16* dst, int idx,
                                         int NT, int N) {
  int j    = idx & 15;
  int lane = (idx >> 4) & 31;
  int f    = idx >> 9;
  int kt   = f / NT, nt = f - kt * NT;
  int n    = nt * 16 + (lane & 15);
  int k    = kt * 32 + (lane >> 4) * 8 + (j & 7) + (j >> 3) * 16;
  float v  = (n < N) ? M[(long)k * N + n] : 0.f;
  dst[idx] = (h16)v;
}

__global__ void pack_weights_kernel(const float* __restrict__ W1,
                                    const float* __restrict__ A1,
                                    const float* __restrict__ W2,
                                    const float* __restrict__ A2,
                                    h16* w1p, h16* a1p, h16* w2p, h16* a2p) {
  int t = blockIdx.x * blockDim.x + threadIdx.x;
  const int n1 = KT1 * NT1 * 512;   // 90112
  const int n2 = n1 + KT1 * NTA * 512;
  const int n3 = n2 + KT2 * NT2 * 512;
  const int n4 = n3 + KTA * 512;
  if      (t < n1) pack_one(W1, w1p, t,       NT1, H1D);
  else if (t < n2) pack_one(A1, a1p, t - n1,  NTA, AHD);
  else if (t < n3) pack_one(W2, w2p, t - n2,  NT2, OUTD);
  else if (t < n4) pack_one(A2, a2p, t - n3,  1,   NH);   // N padded 4->16
}

__global__ void init_kernel(float* pooled, float* denom, unsigned* smx) {
  int t = blockIdx.x * blockDim.x + threadIdx.x;
  if (t < Bg * Nn * OUTD) pooled[t] = 0.f;
  if (t < Bg * Nn * NH) { denom[t] = 0.f; smx[t] = 0u; }  // 0 == fenc(-inf-ish)
}

// ------- main WMMA kernel: 1 wave = 32 edges (2 M-tiles, B reused x2) -------
__global__ __launch_bounds__(128) void edge_mlp_kernel(
    const float* __restrict__ nodes,  const float* __restrict__ edges0,
    const float* __restrict__ globs,  const float* __restrict__ ln_g,
    const float* __restrict__ ln_b,   const float* __restrict__ b1,
    const float* __restrict__ b2,     const float* __restrict__ ab1,
    const float* __restrict__ ab2,
    const int* __restrict__ batch_idx, const int* __restrict__ send_idx,
    const int* __restrict__ recv_idx,
    const h16* __restrict__ w1p, const h16* __restrict__ a1p,
    const h16* __restrict__ w2p, const h16* __restrict__ a2p,
    float* __restrict__ new_edges, float* __restrict__ ascores) {
  __shared__ __align__(16) h16 xt[4][32 * XSTR];   // 94208 B
  __shared__ __align__(16) h16 ht[4][32 * HSTR];   // 67584 B (reused for attn)

  const int lane = threadIdx.x & 31;
  const int w    = threadIdx.x >> 5;
  const int m    = lane & 15;      // A/D row, B/D col selector
  const int hi   = lane >> 4;
  const long e0  = ((long)blockIdx.x * 4 + w) * 32;

  // ---- gather + LayerNorm (two lanes per edge, 2 M-tiles) ----
  for (int mt = 0; mt < 2; ++mt) {
    const long e  = e0 + mt * 16 + m;
    const int  b  = batch_idx[e];
    const float* nS = nodes + ((long)b * Nn + send_idx[e]) * NDIM;
    const float* nR = nodes + ((long)b * Nn + recv_idx[e]) * NDIM;
    h16* xr = &xt[w][(mt * 16 + m) * XSTR];
    float sum = 0.f, ss = 0.f;
    if (hi == 0) {                       // feats [0,160): edges0 + send[0:96)
      const float4* p0 = (const float4*)(edges0 + e * EDIM);
      for (int i = 0; i < 16; ++i) put4(xr + i * 4, p0[i], sum, ss);
      const float4* p1 = (const float4*)nS;
      for (int i = 0; i < 24; ++i) put4(xr + 64 + i * 4, p1[i], sum, ss);
    } else {                             // feats [160,320) + globs
      const float4* p1 = (const float4*)nS;
      for (int i = 24; i < 32; ++i) put4(xr + 64 + i * 4, p1[i], sum, ss);
      const float4* p2 = (const float4*)nR;
      for (int i = 0; i < 32; ++i) put4(xr + 192 + i * 4, p2[i], sum, ss);
      const float4* pg = (const float4*)(globs + b * GDIM);
      for (int i = 0; i < 8; ++i) {      // globals: no LN
        float4 v = pg[i];
        xr[FIN + i * 4 + 0] = (h16)v.x; xr[FIN + i * 4 + 1] = (h16)v.y;
        xr[FIN + i * 4 + 2] = (h16)v.z; xr[FIN + i * 4 + 3] = (h16)v.w;
      }
    }
    sum += __shfl_xor(sum, 16, 32);
    ss  += __shfl_xor(ss, 16, 32);
    const float mu   = sum * (1.f / 320.f);
    const float rstd = rsqrtf(ss * (1.f / 320.f) - mu * mu + 1e-5f);
    const int f0 = hi * 160;
    for (int i = 0; i < 160; ++i) {      // normalize own half in place
      int f = f0 + i;
      float v = (float)xr[f];
      xr[f] = (h16)(((v - mu) * rstd) * ln_g[f] + ln_b[f]);
    }
  }
  __syncthreads();

  const h16* xb = &xt[w][0];

  // ---- GEMM1: H = SiLU(X @ W1 + b1), 32x256; B reused for both M-tiles ----
  for (int nt = 0; nt < NT1; ++nt) {
    v8f acc0 = {}, acc1 = {};
    for (int kt = 0; kt < KT1; ++kt) {
      v16h Bf = ((const v16h*)w1p)[(kt * NT1 + nt) * 32 + lane];
      U16 A;
      const h16* p0 = xb + m * XSTR + kt * 32 + hi * 8;
      A.h[0] = *(const v8h*)p0; A.h[1] = *(const v8h*)(p0 + 16);
      acc0 = wmma16(A.v, Bf, acc0);
      const h16* p1 = p0 + 16 * XSTR;
      A.h[0] = *(const v8h*)p1; A.h[1] = *(const v8h*)(p1 + 16);
      acc1 = wmma16(A.v, Bf, acc1);
    }
    int n = nt * 16 + m;
    float bias = b1[n];
    for (int r = 0; r < 8; ++r) {
      ht[w][(r + 8 * hi) * HSTR + n]      = (h16)silu(acc0[r] + bias);
      ht[w][(16 + r + 8 * hi) * HSTR + n] = (h16)silu(acc1[r] + bias);
    }
  }
  __syncthreads();

  // ---- GEMM2: new_edges = H @ W2 + b2 + edges0 ----
  for (int nt = 0; nt < NT2; ++nt) {
    v8f acc0 = {}, acc1 = {};
    for (int kt = 0; kt < KT2; ++kt) {
      v16h Bf = ((const v16h*)w2p)[(kt * NT2 + nt) * 32 + lane];
      U16 A;
      const h16* p0 = &ht[w][m * HSTR + kt * 32 + hi * 8];
      A.h[0] = *(const v8h*)p0; A.h[1] = *(const v8h*)(p0 + 16);
      acc0 = wmma16(A.v, Bf, acc0);
      const h16* p1 = p0 + 16 * HSTR;
      A.h[0] = *(const v8h*)p1; A.h[1] = *(const v8h*)(p1 + 16);
      acc1 = wmma16(A.v, Bf, acc1);
    }
    int n = nt * 16 + m;
    float bias = b2[n];
    for (int r = 0; r < 8; ++r) {
      long eo0 = e0 + r + 8 * hi;
      long eo1 = eo0 + 16;
      new_edges[eo0 * OUTD + n] = acc0[r] + bias + edges0[eo0 * EDIM + n];
      new_edges[eo1 * OUTD + n] = acc1[r] + bias + edges0[eo1 * EDIM + n];
    }
  }
  __syncthreads();   // ht fully consumed; reuse its space for attn hidden

  h16* atp = &ht[w][0];    // 32 rows x ASTR, aliased into ht buffer

  // ---- GEMM3: Ah = SiLU(X @ A1 + ab1), 32x64 ----
  for (int nt = 0; nt < NTA; ++nt) {
    v8f acc0 = {}, acc1 = {};
    for (int kt = 0; kt < KT1; ++kt) {
      v16h Bf = ((const v16h*)a1p)[(kt * NTA + nt) * 32 + lane];
      U16 A;
      const h16* p0 = xb + m * XSTR + kt * 32 + hi * 8;
      A.h[0] = *(const v8h*)p0; A.h[1] = *(const v8h*)(p0 + 16);
      acc0 = wmma16(A.v, Bf, acc0);
      const h16* p1 = p0 + 16 * XSTR;
      A.h[0] = *(const v8h*)p1; A.h[1] = *(const v8h*)(p1 + 16);
      acc1 = wmma16(A.v, Bf, acc1);
    }
    int n = nt * 16 + m;
    float bias = ab1[n];
    for (int r = 0; r < 8; ++r) {
      atp[(r + 8 * hi) * ASTR + n]      = (h16)silu(acc0[r] + bias);
      atp[(16 + r + 8 * hi) * ASTR + n] = (h16)silu(acc1[r] + bias);
    }
  }
  __syncthreads();

  // ---- GEMM4: scores = (Ah @ A2 + ab2) / 8 ----
  {
    v8f acc0 = {}, acc1 = {};
    for (int kt = 0; kt < KTA; ++kt) {
      v16h Bf = ((const v16h*)a2p)[kt * 32 + lane];
      U16 A;
      const h16* p0 = &atp[m * ASTR + kt * 32 + hi * 8];
      A.h[0] = *(const v8h*)p0; A.h[1] = *(const v8h*)(p0 + 16);
      acc0 = wmma16(A.v, Bf, acc0);
      const h16* p1 = p0 + 16 * ASTR;
      A.h[0] = *(const v8h*)p1; A.h[1] = *(const v8h*)(p1 + 16);
      acc1 = wmma16(A.v, Bf, acc1);
    }
    if (m < NH) {
      float bias = ab2[m];
      for (int r = 0; r < 8; ++r) {
        long eo0 = e0 + r + 8 * hi;
        ascores[eo0 * NH + m]        = (acc0[r] + bias) * 0.125f;
        ascores[(eo0 + 16) * NH + m] = (acc1[r] + bias) * 0.125f;
      }
    }
  }
}

// ---------------- segment softmax + pooled scatter ----------------
__global__ void segmax_kernel(const float* __restrict__ a,
                              const int* __restrict__ bidx,
                              const int* __restrict__ ridx, unsigned* smx) {
  long t = (long)blockIdx.x * blockDim.x + threadIdx.x;
  if (t >= (long)Eg * NH) return;
  long e = t >> 2; int h = (int)(t & 3);
  int seg = bidx[e] * Nn + ridx[e];
  atomicMax(&smx[seg * NH + h], fenc(a[t]));
}

__global__ void expsum_kernel(float* __restrict__ a,
                              const int* __restrict__ bidx,
                              const int* __restrict__ ridx,
                              const unsigned* __restrict__ smx, float* denom) {
  long t = (long)blockIdx.x * blockDim.x + threadIdx.x;
  if (t >= (long)Eg * NH) return;
  long e = t >> 2; int h = (int)(t & 3);
  int seg = bidx[e] * Nn + ridx[e];
  float ae = __expf(a[t] - fdec(smx[seg * NH + h]));
  a[t] = ae;
  atomicAdd(&denom[seg * NH + h], ae);
}

__global__ void pool_kernel(const float* __restrict__ ne,
                            const float* __restrict__ ae,
                            const int* __restrict__ bidx,
                            const int* __restrict__ ridx,
                            const float* __restrict__ denom, float* pooled) {
  long t = (long)blockIdx.x * blockDim.x + threadIdx.x;
  if (t >= (long)Eg * OUTD) return;
  long e = t >> 6; int c = (int)(t & 63); int h = c >> 4;
  int seg = bidx[e] * Nn + ridx[e];
  float attn = ae[e * NH + h] / denom[seg * NH + h];
  atomicAdd(&pooled[(long)seg * OUTD + c], ne[t] * attn);
}

extern "C" void kernel_launch(void* const* d_in, const int* in_sizes, int n_in,
                              void* d_out, int out_size, void* d_ws, size_t ws_size,
                              hipStream_t stream) {
  (void)in_sizes; (void)n_in; (void)out_size; (void)ws_size;
  const float* nodes  = (const float*)d_in[0];
  const float* edges0 = (const float*)d_in[1];
  const float* globs  = (const float*)d_in[2];
  const float* ln_g   = (const float*)d_in[3];
  const float* ln_b   = (const float*)d_in[4];
  const float* W1     = (const float*)d_in[5];
  const float* b1     = (const float*)d_in[6];
  const float* W2     = (const float*)d_in[7];
  const float* b2     = (const float*)d_in[8];
  const float* A1     = (const float*)d_in[9];
  const float* ab1    = (const float*)d_in[10];
  const float* A2     = (const float*)d_in[11];
  const float* ab2    = (const float*)d_in[12];
  const int* bidx     = (const int*)d_in[13];
  const int* sidx     = (const int*)d_in[14];
  const int* ridx     = (const int*)d_in[15];

  char* ws = (char*)d_ws;
  h16*      w1p   = (h16*)(ws + OFF_W1P);
  h16*      a1p   = (h16*)(ws + OFF_A1P);
  h16*      w2p   = (h16*)(ws + OFF_W2P);
  h16*      a2p   = (h16*)(ws + OFF_A2P);
  float*    asc   = (float*)(ws + OFF_ASC);
  unsigned* smx   = (unsigned*)(ws + OFF_SMX);
  float*    denom = (float*)(ws + OFF_DEN);

  float* new_edges = (float*)d_out;
  float* pooled    = new_edges + (long)Eg * OUTD;

  pack_weights_kernel<<<(130048 + 255) / 256, 256, 0, stream>>>(
      W1, A1, W2, A2, w1p, a1p, w2p, a2p);
  init_kernel<<<(Bg * Nn * OUTD + 255) / 256, 256, 0, stream>>>(pooled, denom, smx);
  edge_mlp_kernel<<<Eg / 128, 128, 0, stream>>>(
      nodes, edges0, globs, ln_g, ln_b, b1, b2, ab1, ab2,
      bidx, sidx, ridx, w1p, a1p, w2p, a2p, new_edges, asc);
  segmax_kernel<<<(Eg * NH + 255) / 256, 256, 0, stream>>>(asc, bidx, ridx, smx);
  expsum_kernel<<<(Eg * NH + 255) / 256, 256, 0, stream>>>(asc, bidx, ridx, smx, denom);
  pool_kernel<<<((long)Eg * OUTD + 255) / 256, 256, 0, stream>>>(
      new_edges, asc, bidx, ridx, denom, pooled);
}
